// Linear_6081673691588
// MI455X (gfx1250) — matrix operand order repaired and checked by reading, not compiled, not run on hardware
//
#include <hip/hip_runtime.h>

typedef __attribute__((ext_vector_type(16))) __bf16 v16bf;
typedef __attribute__((ext_vector_type(8)))  __bf16 v8bf;
typedef __attribute__((ext_vector_type(8)))  float  v8f;
typedef __attribute__((ext_vector_type(4)))  unsigned int v4u;
typedef __attribute__((ext_vector_type(8)))  int v8i;
typedef __attribute__((ext_vector_type(4)))  int v4i;

union V16 { v16bf v; v8bf h[2]; };

#define EPSV 1e-5f

#if defined(__HIP_DEVICE_COMPILE__) && __has_builtin(__builtin_amdgcn_tensor_load_to_lds)
#define USE_TDM 1
#else
#define USE_TDM 0
#endif

#if USE_TDM
// Issue one TDM 2D tile load: 32 (K) x 128 (rows) bf16 tile of W[Cout][K] -> LDS,
// with LDS row padding 64B data + 16B pad (matches smW[128][40]).
__device__ __forceinline__ void tdm_load_w_tile(const __bf16* gsrc, unsigned lds_off, int K) {
  unsigned long long ga = (unsigned long long)(uintptr_t)gsrc;
  v4u g0;
  g0[0] = 1u;                                   // count=1, user mode
  g0[1] = lds_off;                              // lds_addr (bytes)
  g0[2] = (unsigned)ga;                         // global_addr[31:0]
  g0[3] = (unsigned)(ga >> 32) | (2u << 30);    // global_addr[56:32] | type=2 (image)
  v8i g1;
  g1[0] = (int)((1u << 16)                      // data_size = 1 -> 2 bytes
              | (1u << 20)                      // pad_enable
              | (3u << 22)                      // pad_interval: 16 DWORDs (64B) per row
              | (3u << 25));                    // pad_amount: 4 DWORDs (16B)
  g1[1] = (int)(((unsigned)K & 0xFFFFu) << 16); // tensor_dim0[15:0] in bits[63:48]
  g1[2] = (int)(128u << 16);                    // tensor_dim1 = 128 rows
  g1[3] = (int)(32u << 16);                     // tile_dim0 = 32 elements
  g1[4] = 128;                                  // tile_dim1 = 128 rows
  g1[5] = K;                                    // tensor_dim0_stride = K elements
  g1[6] = 0;
  g1[7] = 0;
  v4i z4 = (v4i){0, 0, 0, 0};
#if defined(__clang_major__) && __clang_major__ >= 23
  v8i z8 = (v8i){0, 0, 0, 0, 0, 0, 0, 0};
  __builtin_amdgcn_tensor_load_to_lds(g0, g1, z4, z4, z8, 0);
#else
  __builtin_amdgcn_tensor_load_to_lds(g0, g1, z4, z4, 0);
#endif
}

__device__ __forceinline__ void tdm_wait() {
#if __has_builtin(__builtin_amdgcn_s_wait_tensorcnt)
  __builtin_amdgcn_s_wait_tensorcnt(0);
#else
  asm volatile("s_wait_tensorcnt 0x0" ::: "memory");
#endif
}
#endif // USE_TDM

// ---------------- prep: convert W2/W3 to bf16, zero stats ----------------
__global__ void prep_kernel(const float* __restrict__ W2, const float* __restrict__ W3,
                            __bf16* __restrict__ W2b, __bf16* __restrict__ W3b,
                            float* __restrict__ stats) {
  int i = blockIdx.x * blockDim.x + threadIdx.x;
  if (i < 512 * 64)  W2b[i] = (__bf16)W2[i];
  if (i < 512 * 512) W3b[i] = (__bf16)W3[i];
  if (i < 3 * 4 * 512) stats[i] = 0.f;
}

// ---------------- layer 1: [N,3] @ W1[64,3]^T + b1 -> y1 bf16, stats ----------------
__global__ __launch_bounds__(256)
void layer1_kernel(const float* __restrict__ x, const float* __restrict__ W1,
                   const float* __restrict__ b1, __bf16* __restrict__ y1,
                   float* __restrict__ gsum, float* __restrict__ gsq, int N) {
  __shared__ float sW[64 * 3];
  __shared__ float sB[64];
  __shared__ float sSum[64], sSq[64];
  int t = threadIdx.x;
  if (t < 64 * 3) sW[t] = W1[t];
  if (t < 64) { sB[t] = b1[t]; sSum[t] = 0.f; sSq[t] = 0.f; }
  __syncthreads();
  int row = blockIdx.x * blockDim.x + t;
  if (row < N) {
    float x0 = x[row * 3 + 0], x1 = x[row * 3 + 1], x2 = x[row * 3 + 2];
    for (int c = 0; c < 64; ++c) {
      float y = sB[c] + sW[c * 3] * x0 + sW[c * 3 + 1] * x1 + sW[c * 3 + 2] * x2;
      y1[(size_t)row * 64 + c] = (__bf16)y;
      atomicAdd(&sSum[c], y);
      atomicAdd(&sSq[c], y * y);
    }
  }
  __syncthreads();
  if (t < 64) { atomicAdd(&gsum[t], sSum[t]); atomicAdd(&gsq[t], sSq[t]); }
}

// ---------------- finalize BN stats into fused scale/shift ----------------
__global__ void finalize_kernel(const float* __restrict__ gsum, const float* __restrict__ gsq,
                                const float* __restrict__ gamma, const float* __restrict__ beta,
                                float* __restrict__ scale, float* __restrict__ shift,
                                int C, float invN) {
  int c = blockIdx.x * blockDim.x + threadIdx.x;
  if (c >= C) return;
  float mean = gsum[c] * invN;
  float var  = gsq[c] * invN - mean * mean;
  float rstd = rsqrtf(var + EPSV);
  float sc = gamma[c] * rstd;
  scale[c] = sc;
  shift[c] = beta[c] - mean * sc;
}

// ---------------- WMMA GEMM with BN+ReLU applied to the A operand on load ----------------
// Yout[N x Cout] = relu(scaleP*Yprev + shiftP) [N x K] @ Wb[Cout x K]^T + bias
// 256 threads = 8 waves; block tile 128x128; wave tile 32x64 (2x4 wmma accumulators).
// W tile is DMA'd into LDS by the Tensor Data Mover (wave 0), A tile goes through
// VGPRs (BN+ReLU applied) then ds_store.
__global__ __launch_bounds__(256)
void gemm_bn_kernel(const __bf16* __restrict__ Yprev, const __bf16* __restrict__ Wb,
                    const float* __restrict__ bias,
                    const float* __restrict__ scaleP, const float* __restrict__ shiftP,
                    __bf16* __restrict__ Yout,
                    float* __restrict__ gsum, float* __restrict__ gsq,
                    int K, int Cout) {
  __shared__ __bf16 smA[128][40];   // 128 rows x 32 K, padded (+8) for bank spread
  __shared__ __bf16 smW[128][40];   // same layout; TDM pads 16B per 64B row
  __shared__ float sScale[512];
  __shared__ float sShift[512];

  const int t    = threadIdx.x;
  const int lane = t & 31;
  const int wid  = t >> 5;
  const int wm   = (wid & 3) * 32;   // wave M offset within block tile
  const int wn   = (wid >> 2) * 64;  // wave N offset within block tile
  const int blockM = blockIdx.y * 128;
  const int blockN = blockIdx.x * 128;
  const int half = lane >> 4;
  const int l16  = lane & 15;

  for (int i = t; i < K; i += 256) { sScale[i] = scaleP[i]; sShift[i] = shiftP[i]; }

  v8f acc[2][4];
#pragma unroll
  for (int mi = 0; mi < 2; ++mi)
#pragma unroll
    for (int ni = 0; ni < 4; ++ni)
      acc[mi][ni] = (v8f){0.f, 0.f, 0.f, 0.f, 0.f, 0.f, 0.f, 0.f};

  __syncthreads();

  for (int kk = 0; kk < K; kk += 32) {
    __syncthreads();   // previous iteration's fragments fully consumed

#if USE_TDM
    if (wid == 0) {
      tdm_load_w_tile(Wb + (size_t)blockN * K + kk,
                      (unsigned)(uintptr_t)&smW[0][0], K);
    }
#endif

    // Cooperative fill of the 128x32 A tile: 512 v8bf chunks, 2 per thread.
#pragma unroll
    for (int i = 0; i < 2; ++i) {
      int c  = t + i * 256;        // 0..511
      int r  = c >> 2;             // 0..127
      int kc = (c & 3) << 3;       // 0,8,16,24
      const __bf16* ap = Yprev + (size_t)(blockM + r) * K + kk + kc;
      v8bf a = *(const v8bf*)ap;
      if (kk + 32 < K) __builtin_prefetch((const void*)(ap + 32), 0, 0);
      v8bf ao;
#pragma unroll
      for (int j = 0; j < 8; ++j) {
        float f = (float)a[j] * sScale[kk + kc + j] + sShift[kk + kc + j];
        ao[j] = (__bf16)fmaxf(f, 0.f);
      }
      *(v8bf*)&smA[r][kc] = ao;
#if !USE_TDM
      v8bf w = *(const v8bf*)(Wb + (size_t)(blockN + r) * K + kk + kc);
      *(v8bf*)&smW[r][kc] = w;
#endif
    }

#if USE_TDM
    if (wid == 0) tdm_wait();      // TENSORcnt -> 0 before publishing the tile
#endif
    __syncthreads();

    // A fragments: lane<16 -> M=l16, K={0..7,16..23}; lane>=16 -> K={8..15,24..31}
    V16 afrag[2], bfrag[4];
#pragma unroll
    for (int mi = 0; mi < 2; ++mi) {
      int r  = wm + mi * 16 + l16;
      int k0 = half * 8;
      afrag[mi].h[0] = *(const v8bf*)&smA[r][k0];
      afrag[mi].h[1] = *(const v8bf*)&smA[r][k0 + 16];
    }
    // B fragments: lane holds column n=l16, K={0..7,16..23} (half 0) / {8..15,24..31} (half 1)
#pragma unroll
    for (int ni = 0; ni < 4; ++ni) {
      int r  = wn + ni * 16 + l16;
      int k0 = half * 8;
      bfrag[ni].h[0] = *(const v8bf*)&smW[r][k0];
      bfrag[ni].h[1] = *(const v8bf*)&smW[r][k0 + 16];
    }
#pragma unroll
    for (int mi = 0; mi < 2; ++mi)
#pragma unroll
      for (int ni = 0; ni < 4; ++ni)
        acc[mi][ni] = __builtin_amdgcn_wmma_f32_16x16x32_bf16(
            false, afrag[mi].v, false, bfrag[ni].v,
            (short)0, acc[mi][ni], false, false);
  }

  // Epilogue: bias add, raw-y store (bf16), per-channel sum/sumsq atomics.
#pragma unroll
  for (int ni = 0; ni < 4; ++ni) {
    int n = blockN + wn + ni * 16 + l16;
    float bv = bias[n];
    float s = 0.f, sq = 0.f;
#pragma unroll
    for (int mi = 0; mi < 2; ++mi) {
      int rbase = blockM + wm + mi * 16 + half * 8;
#pragma unroll
      for (int v = 0; v < 8; ++v) {
        float y = acc[mi][ni][v] + bv;
        s += y; sq += y * y;
        Yout[(size_t)(rbase + v) * Cout + n] = (__bf16)y;
      }
    }
    atomicAdd(&gsum[n], s);
    atomicAdd(&gsq[n], sq);
  }
}

// ---------------- layer 4: BN3+ReLU then [N,512] @ W4[2,512]^T + b4 -> fp32 out ----------------
__global__ __launch_bounds__(256)
void layer4_kernel(const __bf16* __restrict__ y3,
                   const float* __restrict__ scale3, const float* __restrict__ shift3,
                   const float* __restrict__ W4, const float* __restrict__ b4,
                   float* __restrict__ out, int K, int N) {
  __shared__ float sW0[512], sW1[512], sSc[512], sSh[512];
  int t = threadIdx.x;
  for (int i = t; i < K; i += 256) {
    sW0[i] = W4[i];
    sW1[i] = W4[K + i];
    sSc[i] = scale3[i];
    sSh[i] = shift3[i];
  }
  __syncthreads();
  int row = blockIdx.x * blockDim.x + t;
  if (row >= N) return;
  float a0 = b4[0], a1 = b4[1];
  for (int k = 0; k < K; k += 8) {
    v8bf h = *(const v8bf*)(y3 + (size_t)row * K + k);
#pragma unroll
    for (int j = 0; j < 8; ++j) {
      float f = fmaxf((float)h[j] * sSc[k + j] + sSh[k + j], 0.f);
      a0 += f * sW0[k + j];
      a1 += f * sW1[k + j];
    }
  }
  out[(size_t)row * 2 + 0] = a0;
  out[(size_t)row * 2 + 1] = a1;
}

extern "C" void kernel_launch(void* const* d_in, const int* in_sizes, int n_in,
                              void* d_out, int out_size, void* d_ws, size_t ws_size,
                              hipStream_t stream) {
  // Input order: x1, x2, W1,b1, W2,b2, W3,b3, W4,b4, gamma1,beta1, gamma2,beta2, gamma3,beta3
  const float* x2 = (const float*)d_in[1];
  const float* W1 = (const float*)d_in[2];
  const float* b1 = (const float*)d_in[3];
  const float* W2 = (const float*)d_in[4];
  const float* b2 = (const float*)d_in[5];
  const float* W3 = (const float*)d_in[6];
  const float* b3 = (const float*)d_in[7];
  const float* W4 = (const float*)d_in[8];
  const float* b4 = (const float*)d_in[9];
  const float* gamma1 = (const float*)d_in[10];
  const float* beta1  = (const float*)d_in[11];
  const float* gamma2 = (const float*)d_in[12];
  const float* beta2  = (const float*)d_in[13];
  const float* gamma3 = (const float*)d_in[14];
  const float* beta3  = (const float*)d_in[15];
  float* out = (float*)d_out;

  const int N = in_sizes[1] / 3;           // 262144
  const float invN = 1.0f / (float)N;

  // Workspace carve-up (all offsets 256B-aligned)
  char* ws = (char*)d_ws;
  size_t off = 0;
  __bf16* y1  = (__bf16*)(ws + off); off += (size_t)N * 64 * 2;     // 32 MB
  __bf16* y2  = (__bf16*)(ws + off); off += (size_t)N * 512 * 2;    // 256 MB
  __bf16* y3  = (__bf16*)(ws + off); off += (size_t)N * 512 * 2;    // 256 MB
  __bf16* W2b = (__bf16*)(ws + off); off += (size_t)512 * 64 * 2;
  __bf16* W3b = (__bf16*)(ws + off); off += (size_t)512 * 512 * 2;
  float*  stats = (float*)(ws + off);      // [3 layers][sum|sq|scale|shift][512]
  float* sum0 = stats + 0 * 2048, *sq0 = sum0 + 512, *sc0 = sum0 + 1024, *sh0 = sum0 + 1536;
  float* sum1 = stats + 1 * 2048, *sq1 = sum1 + 512, *sc1 = sum1 + 1024, *sh1 = sum1 + 1536;
  float* sum2 = stats + 2 * 2048, *sq2 = sum2 + 512, *sc2 = sum2 + 1024, *sh2 = sum2 + 1536;

  prep_kernel<<<1024, 256, 0, stream>>>(W2, W3, W2b, W3b, stats);

  layer1_kernel<<<N / 256, 256, 0, stream>>>(x2, W1, b1, y1, sum0, sq0, N);
  finalize_kernel<<<2, 256, 0, stream>>>(sum0, sq0, gamma1, beta1, sc0, sh0, 64, invN);

  gemm_bn_kernel<<<dim3(512 / 128, N / 128), 256, 0, stream>>>(
      y1, W2b, b2, sc0, sh0, y2, sum1, sq1, 64, 512);
  finalize_kernel<<<2, 256, 0, stream>>>(sum1, sq1, gamma2, beta2, sc1, sh1, 512, invN);

  gemm_bn_kernel<<<dim3(512 / 128, N / 128), 256, 0, stream>>>(
      y2, W3b, b3, sc1, sh1, y3, sum2, sq2, 512, 512);
  finalize_kernel<<<2, 256, 0, stream>>>(sum2, sq2, gamma3, beta3, sc2, sh2, 512, invN);

  layer4_kernel<<<N / 256, 256, 0, stream>>>(y3, sc2, sh2, W4, b4, out, 512, N);
}
